// FrequencySparseAttention_24567212933742
// MI455X (gfx1250) — matrix operand order repaired and checked by reading, not compiled, not run on hardware
//
#include <hip/hip_runtime.h>
#include <math.h>

// ---------------------------------------------------------------------------
// FrequencySparseAttention for MI455X (gfx1250, wave32, WMMA)
// B=2, L=2048, D=1024, H=16, Dh=64, C=8, THRESH=0.1
// ---------------------------------------------------------------------------

#define BATCH 2
#define SEQ   2048
#define DMODEL 1024
#define NHEAD 16
#define DHEAD 64
#define NCOEF 8
#define MTOT  (BATCH * SEQ)   // 4096

typedef __attribute__((ext_vector_type(16))) _Float16 v16h;
typedef __attribute__((ext_vector_type(8)))  float    v8f;

union V16H {
    v16h v;
    uint4 q[2];
    _Float16 h[16];
};

// A fragment (16x32 f16): 16B chunks at +0 and +16 halves
__device__ __forceinline__ void loadA(V16H& f, const _Float16* p) {
    f.q[0] = *(const uint4*)(p);
    f.q[1] = *(const uint4*)(p + 16);
}
// B fragment (32x16 f16, operand pre-transposed): 16B chunks at +0 and +16 halves
__device__ __forceinline__ void loadB(V16H& f, const _Float16* p) {
    f.q[0] = *(const uint4*)(p);
    f.q[1] = *(const uint4*)(p + 16);
}

__device__ __forceinline__ v8f v8f_zero() {
    return (v8f){0.f,0.f,0.f,0.f,0.f,0.f,0.f,0.f};
}

// ---------------------------------------------------------------------------
// f32 -> f16 elementwise convert
// ---------------------------------------------------------------------------
__global__ __launch_bounds__(256) void cvt_f16_kernel(const float* __restrict__ in,
                                                      _Float16* __restrict__ out, int n) {
    int i = blockIdx.x * blockDim.x + threadIdx.x;
    if (i < n) out[i] = (_Float16)in[i];
}

// ---------------------------------------------------------------------------
// Weight f32 [K,N] -> f16 transposed [N,K]  (K=N=1024)
// ---------------------------------------------------------------------------
__global__ __launch_bounds__(256) void cvt_wT_kernel(const float* __restrict__ W,
                                                     _Float16* __restrict__ WT) {
    int i = blockIdx.x * blockDim.x + threadIdx.x;   // 0..1M-1
    int n = i & (DMODEL - 1);
    int k = i >> 10;
    WT[(size_t)n * DMODEL + k] = (_Float16)W[(size_t)k * DMODEL + n];
}

// ---------------------------------------------------------------------------
// gate[b*L + l] = sigmoid((mean_c |E| - 0.1) * 10)
// ---------------------------------------------------------------------------
__global__ __launch_bounds__(256) void gate_kernel(const float* __restrict__ E,
                                                   float* __restrict__ gate, int n) {
    int i = blockIdx.x * blockDim.x + threadIdx.x;
    if (i >= n) return;
    float s = 0.f;
#pragma unroll
    for (int c = 0; c < NCOEF; ++c) s += fabsf(E[i * NCOEF + c]);
    float e = s * (1.0f / NCOEF);
    gate[i] = 1.0f / (1.0f + __expf(-(e - 0.1f) * 10.0f));
}

// ---------------------------------------------------------------------------
// WMMA GEMM:  Out = A[M,1024](f16,rm) x WT[N,K](f16, B pre-transposed) + bias.
// Block = 128 threads = 4 waves; block tile 128(M)x64(N); each wave owns a
// 32x64 tile (8 accumulators) with double-buffered A/B fragments:
// iteration i issues loads for i+1, then runs 8 WMMAs on registers loaded a
// full iteration earlier -> partial loadcnt waits instead of full drains.
//   mode 0: write f16, reshape to [B,H,L,Dh]      (Q, K)
//   mode 1: write f16, reshape to [B,H,Dh,L]      (V transposed)
//   mode 2: write f32, row-major [M,N]            (final output)
// ---------------------------------------------------------------------------
__global__ __launch_bounds__(128) void gemm_wmma_kernel(const _Float16* __restrict__ A,
                                                        const _Float16* __restrict__ WT,
                                                        const float* __restrict__ bias,
                                                        void* __restrict__ out, int mode) {
    const int wave = threadIdx.x >> 5;
    const int lane = threadIdx.x & 31;
    const int lh = lane & 15;
    const int kh = lane >> 4;

    const int mbase = blockIdx.y * 128 + wave * 32;
    const int nbase = blockIdx.x * 64;

    v8f acc[2][4];
#pragma unroll
    for (int g = 0; g < 2; ++g)
#pragma unroll
        for (int t = 0; t < 4; ++t) acc[g][t] = v8f_zero();

    const _Float16* ap0 = A + (size_t)(mbase + lh) * DMODEL + kh * 8;
    const _Float16* ap1 = A + (size_t)(mbase + 16 + lh) * DMODEL + kh * 8;
    const _Float16* bp[4];
#pragma unroll
    for (int t = 0; t < 4; ++t)
        bp[t] = WT + (size_t)(nbase + t * 16 + lh) * DMODEL + kh * 8;

    V16H a[2][2], b[2][4];
    loadA(a[0][0], ap0);
    loadA(a[0][1], ap1);
#pragma unroll
    for (int t = 0; t < 4; ++t) loadB(b[0][t], bp[t]);

    int buf = 0;
    for (int kk = 32; kk < DMODEL; kk += 32) {
        const int nb = buf ^ 1;
        // prefetch next K-step fragments
        loadA(a[nb][0], ap0 + kk);
        loadA(a[nb][1], ap1 + kk);
#pragma unroll
        for (int t = 0; t < 4; ++t) loadB(b[nb][t], bp[t] + kk);
        // compute on fragments issued one iteration ago
#pragma unroll
        for (int t = 0; t < 4; ++t)
            acc[0][t] = __builtin_amdgcn_wmma_f32_16x16x32_f16(
                false, a[buf][0].v, false, b[buf][t].v, (short)0, acc[0][t], false, false);
#pragma unroll
        for (int t = 0; t < 4; ++t)
            acc[1][t] = __builtin_amdgcn_wmma_f32_16x16x32_f16(
                false, a[buf][1].v, false, b[buf][t].v, (short)0, acc[1][t], false, false);
        buf = nb;
    }
    // epilogue K-step
#pragma unroll
    for (int t = 0; t < 4; ++t)
        acc[0][t] = __builtin_amdgcn_wmma_f32_16x16x32_f16(
            false, a[buf][0].v, false, b[buf][t].v, (short)0, acc[0][t], false, false);
#pragma unroll
    for (int t = 0; t < 4; ++t)
        acc[1][t] = __builtin_amdgcn_wmma_f32_16x16x32_f16(
            false, a[buf][1].v, false, b[buf][t].v, (short)0, acc[1][t], false, false);

#pragma unroll
    for (int g = 0; g < 2; ++g) {
#pragma unroll
        for (int t = 0; t < 4; ++t) {
            const int col = nbase + t * 16 + lh;
            const float bv = bias[col];
#pragma unroll
            for (int r = 0; r < 8; ++r) {
                const int m = mbase + g * 16 + kh * 8 + r;
                const float val = acc[g][t][r] + bv;
                if (mode == 2) {
                    ((float*)out)[(size_t)m * DMODEL + col] = val;
                } else {
                    const int bI = m >> 11;          // /SEQ
                    const int l  = m & (SEQ - 1);
                    const int h  = col >> 6;         // /DHEAD
                    const int dh = col & (DHEAD - 1);
                    if (mode == 0) {
                        ((_Float16*)out)[(((size_t)(bI * NHEAD + h)) * SEQ + l) * DHEAD + dh] =
                            (_Float16)val;
                    } else { // mode 1: V transposed [B,H,Dh,L]
                        ((_Float16*)out)[(((size_t)(bI * NHEAD + h)) * DHEAD + dh) * SEQ + l] =
                            (_Float16)val;
                    }
                }
            }
        }
    }
}

// ---------------------------------------------------------------------------
// Flash-style gated attention.
// grid = (L/64, B*H); block = 128 threads = 4 waves; wave -> 16 q-rows.
// Q,K: [B,H,L,Dh] f16.  VT: [B,H,Dh,L] f16.  gate: [B,L] f32. O: [B,L,D] f16.
// out_q = sum_k softmax(s)_k * gate_k * V_k   (denominator ungated).
// K fragments double-buffered across key tiles; V fragments issued at loop
// top so all global loads overlap the softmax VALU work.
// ---------------------------------------------------------------------------
__global__ __launch_bounds__(128) void attn_kernel(const _Float16* __restrict__ Q,
                                                   const _Float16* __restrict__ K,
                                                   const _Float16* __restrict__ VT,
                                                   const float* __restrict__ gate,
                                                   _Float16* __restrict__ O) {
    __shared__ _Float16 lds[4][16][32];   // per-wave P tile (16q x 32k)

    const int wave = threadIdx.x >> 5;
    const int lane = threadIdx.x & 31;
    const int lh = lane & 15;
    const int kh = lane >> 4;

    const int bh = blockIdx.y;            // 0..31
    const int b  = bh >> 4;
    const int h  = bh & (NHEAD - 1);
    const int qbase = blockIdx.x * 64 + wave * 16;

    const _Float16* Qb = Q  + (size_t)bh * SEQ * DHEAD;
    const _Float16* Kb = K  + (size_t)bh * SEQ * DHEAD;
    const _Float16* Vb = VT + (size_t)bh * DHEAD * SEQ;
    const float*    gb = gate + (size_t)b * SEQ;

    // Q fragments: A-layout, two d-chunks (Dh=64 -> 2 x 16x32)
    V16H qf[2];
    const int qrow = qbase + lh;
#pragma unroll
    for (int c = 0; c < 2; ++c)
        loadA(qf[c], Qb + (size_t)qrow * DHEAD + c * 32 + kh * 8);

    float m_[8], l_[8];
#pragma unroll
    for (int r = 0; r < 8; ++r) { m_[r] = -3.0e38f; l_[r] = 0.0f; }
    v8f acc[4];
#pragma unroll
    for (int t = 0; t < 4; ++t) acc[t] = v8f_zero();

    const float scale = 0.125f;   // 1/sqrt(64)

    // K fragment double buffer: [buf][key-subtile][d-chunk]
    V16H kf[2][2][2];
#pragma unroll
    for (int n = 0; n < 2; ++n)
#pragma unroll
        for (int c = 0; c < 2; ++c)
            loadB(kf[0][n][c], Kb + (size_t)(n * 16 + lh) * DHEAD + c * 32 + kh * 8);

    int buf = 0;
    for (int kt = 0; kt < SEQ; kt += 32) {
        // ---- issue V fragment loads for this tile early --------------------
        V16H vf[4];
#pragma unroll
        for (int t = 0; t < 4; ++t)
            loadB(vf[t], Vb + (size_t)(t * 16 + lh) * SEQ + kt + kh * 8);
        // ---- prefetch next tile's K fragments ------------------------------
        const int nb = buf ^ 1;
        if (kt + 32 < SEQ) {
            const int knext = kt + 32;
#pragma unroll
            for (int n = 0; n < 2; ++n)
#pragma unroll
                for (int c = 0; c < 2; ++c)
                    loadB(kf[nb][n][c],
                          Kb + (size_t)(knext + n * 16 + lh) * DHEAD + c * 32 + kh * 8);
        }
        // ---- scores S[16q, 32k] --------------------------------------------
        v8f s[2];
#pragma unroll
        for (int n = 0; n < 2; ++n) {
            s[n] = v8f_zero();
#pragma unroll
            for (int c = 0; c < 2; ++c)
                s[n] = __builtin_amdgcn_wmma_f32_16x16x32_f16(
                    false, qf[c].v, false, kf[buf][n][c].v, (short)0, s[n], false, false);
        }
        // ---- online softmax (rows live in lane halves) ---------------------
        float tmax[8];
#pragma unroll
        for (int r = 0; r < 8; ++r) {
            s[0][r] *= scale; s[1][r] *= scale;
            float v = fmaxf(s[0][r], s[1][r]);
#pragma unroll
            for (int msk = 1; msk < 16; msk <<= 1) v = fmaxf(v, __shfl_xor(v, msk, 32));
            tmax[r] = v;
        }
        const float g0 = gb[kt + lh];
        const float g1 = gb[kt + 16 + lh];
#pragma unroll
        for (int r = 0; r < 8; ++r) {
            const float mnew = fmaxf(m_[r], tmax[r]);
            const float alpha = __expf(m_[r] - mnew);
            m_[r] = mnew;
            const float p0 = __expf(s[0][r] - mnew);
            const float p1 = __expf(s[1][r] - mnew);
            float prs = p0 + p1;
#pragma unroll
            for (int msk = 1; msk < 16; msk <<= 1) prs += __shfl_xor(prs, msk, 32);
            l_[r] = l_[r] * alpha + prs;
#pragma unroll
            for (int t = 0; t < 4; ++t) acc[t][r] *= alpha;
            // gated probabilities -> LDS (P tile, A-layout source)
            const int row = kh * 8 + r;
            lds[wave][row][lh]      = (_Float16)(p0 * g0);
            lds[wave][row][16 + lh] = (_Float16)(p1 * g1);
        }
        asm volatile("s_wait_dscnt 0" ::: "memory");

        // ---- P as A fragment (16x32 over keys) -----------------------------
        V16H pf;
        {
            const _Float16* lp = &lds[wave][lh][kh * 8];
            pf.q[0] = *(const uint4*)(lp);
            pf.q[1] = *(const uint4*)(lp + 16);
        }
        // ---- O += P x V : 4 d-tiles (vf issued at loop top) ----------------
#pragma unroll
        for (int t = 0; t < 4; ++t)
            acc[t] = __builtin_amdgcn_wmma_f32_16x16x32_f16(
                false, pf.v, false, vf[t].v, (short)0, acc[t], false, false);
        asm volatile("s_wait_dscnt 0" ::: "memory");
        buf = nb;
    }

    // ---- finalize: divide by ungated softmax sum, write [B,L,D] f16 --------
#pragma unroll
    for (int t = 0; t < 4; ++t) {
#pragma unroll
        for (int r = 0; r < 8; ++r) {
            const int qr = qbase + kh * 8 + r;
            const int col = h * DHEAD + t * 16 + lh;
            const float inv = 1.0f / l_[r];
            O[((size_t)b * SEQ + qr) * DMODEL + col] = (_Float16)(acc[t][r] * inv);
        }
    }
}

// ---------------------------------------------------------------------------
// Host launcher
// ---------------------------------------------------------------------------
extern "C" void kernel_launch(void* const* d_in, const int* in_sizes, int n_in,
                              void* d_out, int out_size, void* d_ws, size_t ws_size,
                              hipStream_t stream) {
    (void)in_sizes; (void)n_in; (void)out_size; (void)ws_size;

    const float* q_x = (const float*)d_in[0];
    const float* k_x = (const float*)d_in[1];
    const float* v_x = (const float*)d_in[2];
    const float* E   = (const float*)d_in[3];
    const float* Wq  = (const float*)d_in[4];
    const float* bq  = (const float*)d_in[5];
    const float* Wk  = (const float*)d_in[6];
    const float* bk  = (const float*)d_in[7];
    const float* Wv  = (const float*)d_in[8];
    const float* bv  = (const float*)d_in[9];
    const float* Wo  = (const float*)d_in[10];
    const float* bo  = (const float*)d_in[11];
    float* out = (float*)d_out;

    char* ws = (char*)d_ws;
    const size_t MB = 1024 * 1024;
    _Float16* Aq  = (_Float16*)(ws + 0 * MB);    // 8 MB each (4M f16)
    _Float16* Ak  = (_Float16*)(ws + 8 * MB);
    _Float16* Av  = (_Float16*)(ws + 16 * MB);
    _Float16* WqT = (_Float16*)(ws + 24 * MB);   // 2 MB each
    _Float16* WkT = (_Float16*)(ws + 26 * MB);
    _Float16* WvT = (_Float16*)(ws + 28 * MB);
    _Float16* WoT = (_Float16*)(ws + 30 * MB);
    _Float16* Q16 = (_Float16*)(ws + 32 * MB);   // 8 MB each
    _Float16* K16 = (_Float16*)(ws + 40 * MB);
    _Float16* V16 = (_Float16*)(ws + 48 * MB);
    _Float16* O16 = (_Float16*)(ws + 56 * MB);
    float*    gte = (float*)   (ws + 64 * MB);   // 16 KB

    const int NACT = MTOT * DMODEL;   // 4M
    const int NW   = DMODEL * DMODEL; // 1M

    // 1) convert activations + transpose-convert weights
    cvt_f16_kernel<<<NACT / 256, 256, 0, stream>>>(q_x, Aq, NACT);
    cvt_f16_kernel<<<NACT / 256, 256, 0, stream>>>(k_x, Ak, NACT);
    cvt_f16_kernel<<<NACT / 256, 256, 0, stream>>>(v_x, Av, NACT);
    cvt_wT_kernel<<<NW / 256, 256, 0, stream>>>(Wq, WqT);
    cvt_wT_kernel<<<NW / 256, 256, 0, stream>>>(Wk, WkT);
    cvt_wT_kernel<<<NW / 256, 256, 0, stream>>>(Wv, WvT);
    cvt_wT_kernel<<<NW / 256, 256, 0, stream>>>(Wo, WoT);

    // 2) key gate
    gate_kernel<<<(BATCH * SEQ) / 256, 256, 0, stream>>>(E, gte, BATCH * SEQ);

    // 3) projections (WMMA GEMMs)
    dim3 ggrid(DMODEL / 64, MTOT / 128);
    gemm_wmma_kernel<<<ggrid, 128, 0, stream>>>(Aq, WqT, bq, (void*)Q16, 0);
    gemm_wmma_kernel<<<ggrid, 128, 0, stream>>>(Ak, WkT, bk, (void*)K16, 0);
    gemm_wmma_kernel<<<ggrid, 128, 0, stream>>>(Av, WvT, bv, (void*)V16, 1);

    // 4) gated flash attention
    dim3 agrid(SEQ / 64, BATCH * NHEAD);
    attn_kernel<<<agrid, 128, 0, stream>>>(Q16, K16, V16, gte, O16);

    // 5) output projection -> f32 d_out
    gemm_wmma_kernel<<<ggrid, 128, 0, stream>>>(O16, WoT, bo, (void*)out, 2);
}
